// SupervisedGraphSage_56736517980222
// MI455X (gfx1250) — matrix operand order repaired and checked by reading, not compiled, not run on hardware
//
#include <hip/hip_runtime.h>
#include <hip/hip_bf16.h>
#include <math.h>

// ---------------- problem constants ----------------
#define N_NODES   200000
#define N_FEATS   602
#define KPAD1     608        // 602 padded to 19 * 32
#define EMBED     128
#define NUM_CLS   41
#define NSAMP     5
#define BATCH     8192
#define M1        (BATCH * 6)      // 49152 layer-1 rows

// ---------------- vector types ----------------
typedef float        v8f    __attribute__((ext_vector_type(8)));
typedef __bf16       v16bf  __attribute__((ext_vector_type(16)));
typedef unsigned int u32x4  __attribute__((ext_vector_type(4)));
typedef int          v4i_vs __attribute__((vector_size(16)));   // matches builtin proto

union Frag16 { u32x4 u[2]; v16bf v; };

// ---------------- CDNA5 async global->LDS path (probe via __has_builtin) ----
#if __has_builtin(__builtin_amdgcn_global_load_async_to_lds_b128) && \
    __has_builtin(__builtin_amdgcn_s_wait_asynccnt)
#define USE_ASYNC_LDS 1
#else
#define USE_ASYNC_LDS 0
#endif

__device__ __forceinline__ void cp_b128(void* lds, const void* gmem) {
#if USE_ASYNC_LDS
    __builtin_amdgcn_global_load_async_to_lds_b128(
        (__attribute__((address_space(1))) v4i_vs*)gmem,
        (__attribute__((address_space(3))) v4i_vs*)lds,
        /*offset=*/0, /*cpol=*/0);
#else
    *reinterpret_cast<u32x4*>(lds) = *reinterpret_cast<const u32x4*>(gmem);
#endif
}

__device__ __forceinline__ void wait_cp() {
#if USE_ASYNC_LDS
    __builtin_amdgcn_s_wait_asynccnt(0);
#endif
}

// ---------------- workspace layout (bytes) ----------------
#define OFF_H0   ((size_t)0)                                  // bf16 [M1, 608]
#define OFF_H1   (OFF_H0 + (size_t)M1 * KPAD1 * 2)            // f32  [M1, 128]
#define OFF_H2I  (OFF_H1 + (size_t)M1 * EMBED * 4)            // bf16 [B, 128]
#define OFF_H2   (OFF_H2I + (size_t)BATCH * EMBED * 2)        // bf16 [B, 128]
#define OFF_W1B  (OFF_H2 + (size_t)BATCH * EMBED * 2)         // bf16 [128, 608]
#define OFF_W2B  (OFF_W1B + (size_t)EMBED * KPAD1 * 2)        // bf16 [128, 128]

// ---------------- weight downcast / pad ----------------
__global__ void cvt_w1_pad(const float* __restrict__ W1, __bf16* __restrict__ W1b) {
    int idx = blockIdx.x * blockDim.x + threadIdx.x;     // EMBED * KPAD1
    if (idx >= EMBED * KPAD1) return;
    int n = idx / KPAD1, k = idx - n * KPAD1;
    W1b[idx] = (k < N_FEATS) ? (__bf16)W1[(size_t)n * N_FEATS + k] : (__bf16)0.0f;
}

__global__ void cvt_f32_bf16(const float* __restrict__ src, __bf16* __restrict__ dst, int n) {
    int idx = blockIdx.x * blockDim.x + threadIdx.x;
    if (idx < n) dst[idx] = (__bf16)src[idx];
}

// ---------------- fused 2-hop gather + mean + bf16 downcast ----------------
// one block per layer-1 row m; h0[m, :] = mean of 6 feature rows (self + 5 neighbors)
__global__ __launch_bounds__(128) void gather_mean_bf16(
        const float* __restrict__ features, const int* __restrict__ neigh,
        const int* __restrict__ nodes, __bf16* __restrict__ h0) {
    const int m = blockIdx.x;                 // 0..M1-1
    const int b = m / 6;
    const int j = m - b * 6;
    const int nb = nodes[b];                                  // uniform -> SGPR
    const int nm = (j == 0) ? nb : neigh[nb * NSAMP + (j - 1)];
    int src[6];
    src[0] = nm;
#pragma unroll
    for (int s = 0; s < NSAMP; ++s) src[s + 1] = neigh[nm * NSAMP + s];

    const int tid = threadIdx.x;
    // prefetch the six gathered rows (global_prefetch_b8) to cover gather latency
    if (tid < 6) {
        const float* p = features + (size_t)src[tid] * N_FEATS;
        __builtin_prefetch(p, 0, 3);
        __builtin_prefetch(p + 301, 0, 3);
    }

    for (int c = tid; c < KPAD1; c += 128) {
        __bf16 v;
        if (c < N_FEATS) {
            float acc = 0.0f;
#pragma unroll
            for (int s = 0; s < 6; ++s)
                acc += features[(size_t)src[s] * N_FEATS + c];
            v = (__bf16)(acc * (1.0f / 6.0f));
        } else {
            v = (__bf16)0.0f;
        }
        h0[(size_t)m * KPAD1 + c] = v;
    }
}

// ---------------- WMMA GEMM: out = relu(A @ B^T) ----------------
// A: [M, Kp] bf16 row-major; B: [128, Kp] bf16 row-major (row = output column)
// block = 128 threads (4 waves); tile = 64 (M) x 128 (N); Kp % 32 == 0
// double-buffered LDS, async global->LDS copies (ASYNCcnt pipeline)
template <bool OUT_BF16>
__global__ __launch_bounds__(128) void gemm_relu_64x128(
        const __bf16* __restrict__ A, const __bf16* __restrict__ B,
        void* __restrict__ out, int Kp) {
    __shared__ u32x4 As[2][64 * 5];    // 64 rows x 32 bf16, padded to 80B/row
    __shared__ u32x4 Bs[2][128 * 5];   // 128 rows x 32 bf16, padded

    const int tid  = threadIdx.x;
    const int lane = tid & 31;
    const int wave = tid >> 5;
    const int m0   = blockIdx.x * 64;
    const int lrow = lane & 15;
    const int lhi  = lane >> 4;

    // staging assignment: thread -> (row = tid/4, 16B-chunk q = tid%4), fully static
    const int ra = tid >> 2;           // 0..31
    const int q  = tid & 3;            // 0..3
    const __bf16* gA0 = A + (size_t)(m0 + ra)      * Kp + q * 8;
    const __bf16* gA1 = A + (size_t)(m0 + ra + 32) * Kp + q * 8;
    const __bf16* gB0 = B + (size_t)(ra)           * Kp + q * 8;
    const __bf16* gB1 = B + (size_t)(ra + 32)      * Kp + q * 8;
    const __bf16* gB2 = B + (size_t)(ra + 64)      * Kp + q * 8;
    const __bf16* gB3 = B + (size_t)(ra + 96)      * Kp + q * 8;

    auto issue = [&](int bufi, int kc) {
        const int ko = kc * 32;
        cp_b128(&As[bufi][ra * 5 + q],        gA0 + ko);
        cp_b128(&As[bufi][(ra + 32) * 5 + q], gA1 + ko);
        cp_b128(&Bs[bufi][ra * 5 + q],        gB0 + ko);
        cp_b128(&Bs[bufi][(ra + 32) * 5 + q], gB1 + ko);
        cp_b128(&Bs[bufi][(ra + 64) * 5 + q], gB2 + ko);
        cp_b128(&Bs[bufi][(ra + 96) * 5 + q], gB3 + ko);
    };

    v8f acc[8];
#pragma unroll
    for (int t = 0; t < 8; ++t) acc[t] = (v8f)0.0f;

    const int kchunks = Kp >> 5;
    issue(0, 0);
    int buf = 0;
    for (int kc = 0; kc < kchunks; ++kc) {
        wait_cp();                      // s_wait_asynccnt 0 (this wave's copies done)
        __syncthreads();                // all waves' copies for `buf` visible
        if (kc + 1 < kchunks) issue(buf ^ 1, kc + 1);   // prefetch next chunk

        // A fragment: 16x32, lanes 0-15 get K[8*lhi..], K[16+8*lhi..]
        Frag16 a;
        a.u[0] = As[buf][(wave * 16 + lrow) * 5 + lhi];
        a.u[1] = As[buf][(wave * 16 + lrow) * 5 + 2 + lhi];

        // preload all 8 B fragments, then chain the WMMAs
        Frag16 bfr[8];
#pragma unroll
        for (int t = 0; t < 8; ++t) {
            bfr[t].u[0] = Bs[buf][(t * 16 + lrow) * 5 + lhi];
            bfr[t].u[1] = Bs[buf][(t * 16 + lrow) * 5 + 2 + lhi];
        }
#pragma unroll
        for (int t = 0; t < 8; ++t) {
            acc[t] = __builtin_amdgcn_wmma_f32_16x16x32_bf16(
                false, a.v, false, bfr[t].v, (short)0, acc[t], false, false);
        }
        buf ^= 1;
    }

    // epilogue: C/D layout -> VGPR r: M = r (+8 for lanes 16-31), N = lane&15
    const int orow0 = m0 + wave * 16 + lhi * 8;
#pragma unroll
    for (int t = 0; t < 8; ++t) {
        const int col = t * 16 + lrow;
#pragma unroll
        for (int r = 0; r < 8; ++r) {
            float v = acc[t][r];
            v = v > 0.0f ? v : 0.0f;
            const size_t o = (size_t)(orow0 + r) * EMBED + col;
            if (OUT_BF16) ((__bf16*)out)[o] = (__bf16)v;
            else          ((float*)out)[o]  = v;
        }
    }
}

// ---------------- layer-2 mean over 6 rows, downcast to bf16 ----------------
__global__ __launch_bounds__(128) void mean6_bf16(
        const float* __restrict__ h1, __bf16* __restrict__ h2in) {
    const int b = blockIdx.x, c = threadIdx.x;
    float acc = 0.0f;
#pragma unroll
    for (int j = 0; j < 6; ++j)
        acc += h1[(size_t)(b * 6 + j) * EMBED + c];
    h2in[(size_t)b * EMBED + c] = (__bf16)(acc * (1.0f / 6.0f));
}

// ---------------- classifier: sigmoid(h2 @ Wc^T), fp32 weights ----------------
__global__ __launch_bounds__(64) void classifier_sigmoid(
        const __bf16* __restrict__ h2, const float* __restrict__ Wc,
        float* __restrict__ out) {
    __shared__ float row[EMBED];
    const int b = blockIdx.x, t = threadIdx.x;
    row[t]      = (float)h2[(size_t)b * EMBED + t];
    row[t + 64] = (float)h2[(size_t)b * EMBED + t + 64];
    __syncthreads();
    if (t < NUM_CLS) {
        float acc = 0.0f;
#pragma unroll 8
        for (int k = 0; k < EMBED; ++k)
            acc += row[k] * Wc[(size_t)t * EMBED + k];
        out[(size_t)b * NUM_CLS + t] = 1.0f / (1.0f + __expf(-acc));
    }
}

// ---------------- launch ----------------
extern "C" void kernel_launch(void* const* d_in, const int* in_sizes, int n_in,
                              void* d_out, int out_size, void* d_ws, size_t ws_size,
                              hipStream_t stream) {
    const float* features = (const float*)d_in[0];
    const int*   neigh    = (const int*)d_in[1];
    const int*   nodes    = (const int*)d_in[2];
    const float* W1       = (const float*)d_in[3];
    const float* W2       = (const float*)d_in[4];
    const float* Wc       = (const float*)d_in[5];

    char* ws = (char*)d_ws;
    __bf16* h0   = (__bf16*)(ws + OFF_H0);
    float*  h1   = (float*)(ws + OFF_H1);
    __bf16* h2in = (__bf16*)(ws + OFF_H2I);
    __bf16* h2   = (__bf16*)(ws + OFF_H2);
    __bf16* W1b  = (__bf16*)(ws + OFF_W1B);
    __bf16* W2b  = (__bf16*)(ws + OFF_W2B);

    cvt_w1_pad<<<(EMBED * KPAD1 + 255) / 256, 256, 0, stream>>>(W1, W1b);
    cvt_f32_bf16<<<(EMBED * EMBED + 255) / 256, 256, 0, stream>>>(W2, W2b, EMBED * EMBED);

    gather_mean_bf16<<<M1, 128, 0, stream>>>(features, neigh, nodes, h0);

    gemm_relu_64x128<false><<<M1 / 64, 128, 0, stream>>>(h0, W1b, (void*)h1, KPAD1);

    mean6_bf16<<<BATCH, 128, 0, stream>>>(h1, h2in);

    gemm_relu_64x128<true><<<BATCH / 64, 128, 0, stream>>>(h2in, W2b, (void*)h2, EMBED);

    classifier_sigmoid<<<BATCH, 64, 0, stream>>>(h2, Wc, (float*)d_out);
}
